// Bidomain_Cross_Attention_8100308320820
// MI455X (gfx1250) — matrix operand order repaired and checked by reading, compiled, not run on hardware
//
#include <hip/hip_runtime.h>
#include <hip/hip_bf16.h>

typedef _Float16 f16;
typedef f16   v16h __attribute__((ext_vector_type(16)));
typedef float v8f  __attribute__((ext_vector_type(8)));
typedef float v2f  __attribute__((ext_vector_type(2)));
typedef float v4f  __attribute__((ext_vector_type(4)));

#define HW_   65536
#define BN_   4

// ---------------------------------------------------------------------------
// Pointwise (1x1 conv) GEMM, f16 WMMA + f32 accumulate, LDS-staged B operand.
// out[b, co, p] = bias[co] + sum_ci W[b*wBS + co*Cin + ci] * in[b*inBS + ci*HW + p]
// Cin in {64,128}, Cout in {64,128}; numCo=Cout/16 must divide 8.
// One workgroup (8 waves) stages a Cin x PX input tile in LDS once; wave w
// computes co-tile (w % numCo) on pixel sub-tile (w / numCo).
// ---------------------------------------------------------------------------
__global__ __launch_bounds__(256)
void pw_gemm_wmma(const float* __restrict__ in, const float* __restrict__ W,
                  const float* __restrict__ bias, float* __restrict__ out,
                  int Cin, int Cout, long inBS, long outBS, long wBS)
{
    __shared__ f16 ldsB[4608];            // [px][ci], row stride LDW (<= 9.2 KB)

    const int tid  = threadIdx.x;
    const int lane = tid & 31;
    const int ln   = lane & 15;
    const int hi   = lane >> 4;
    const int wv   = tid >> 5;            // wave in workgroup (0..7)

    const int numCo = Cout >> 4;
    const int tPW   = 8 / numCo;          // pixel sub-tiles per workgroup
    const int PX    = 16 * tPW;           // pixels staged per workgroup
    const int LDW   = Cin + 16;           // padded LDS row stride (halves, %16==0)
    const int nkc   = Cin >> 5;

    const int cot = wv % numCo;
    const int sub = wv / numCo;
    const int m   = (cot << 4) + ln;      // weight row (M) for this lane

    const int numPG = HW_ / PX;
    const long totalG = (long)BN_ * numPG;

    for (long g = blockIdx.x; g < totalG; g += gridDim.x) {
        const int ptg   = (int)(g % numPG);
        const int b     = (int)(g / numPG);
        const int pbase = ptg * PX;
        const float* inB  = in + (long)b * inBS;
        const float* wrow = W  + (long)b * wBS + (long)m * Cin;

        // cooperative, coalesced fill: LDS[px][ci] = (f16) in[ci, pbase+px]
        const int nElem = Cin * PX;
        for (int i = tid; i < nElem; i += 256) {
            int ci = i / PX, px = i % PX;
            ldsB[px * LDW + ci] = (f16)inB[(long)ci * HW_ + pbase + px];
        }
        __syncthreads();

        v8f c = {};
        const f16* bTile = &ldsB[(sub * 16 + ln) * LDW];
        for (int kc = 0; kc < nkc; ++kc) {
            const int kA = kc * 32 + (hi ? 16 : 0);
            const int kB = kc * 32 + (hi ? 16 : 0);

            // A: one contiguous 16-float run -> 4x float4 loads (b128, cached)
            const v4f* wr = (const v4f*)(wrow + kA);
            v4f w0 = wr[0], w1 = wr[1], w2 = wr[2], w3 = wr[3];
            v16h a;
#pragma unroll
            for (int e = 0; e < 4; ++e) {
                a[e]      = (f16)w0[e];
                a[e + 4]  = (f16)w1[e];
                a[e + 8]  = (f16)w2[e];
                a[e + 12] = (f16)w3[e];
            }

            // B fragment: rows kB..kB+15 at this lane's pixel = one 32B LDS read
            v16h bbv = *(const v16h*)(bTile + kB);

            c = __builtin_amdgcn_wmma_f32_16x16x32_f16(false, a, false, bbv,
                                                       (short)0, c, false, false);
        }

        // hoisted bias (single uniform branch, 8 cached loads)
        float bv[8];
        if (bias) {
            const float* bp = bias + (cot << 4) + (hi ? 8 : 0);
#pragma unroll
            for (int r = 0; r < 8; ++r) bv[r] = bp[r];
        } else {
#pragma unroll
            for (int r = 0; r < 8; ++r) bv[r] = 0.f;
        }

        float* orow = out + (long)b * outBS
                    + (long)((cot << 4) + (hi ? 8 : 0)) * HW_
                    + pbase + sub * 16 + ln;
#pragma unroll
        for (int r = 0; r < 8; ++r)
            orow[r * HW_] = c[r] + bv[r];

        __syncthreads();                  // LDS reused next iteration
    }
}

// ---------------------------------------------------------------------------
// Depthwise 5x5, pad 2. One block = 256 pixels of one (b,c) row.
// ---------------------------------------------------------------------------
__global__ __launch_bounds__(256)
void dw_conv5(const float* __restrict__ in, const float* __restrict__ w,
              const float* __restrict__ bias, float* __restrict__ out, int Cn)
{
    const int chunk = blockIdx.x & 255;       // row index y
    const int bc    = blockIdx.x >> 8;        // (b*Cn + c)
    const int c     = bc % Cn;

    __shared__ float wl[26];
    if (threadIdx.x < 25) wl[threadIdx.x] = w[c * 25 + threadIdx.x];
    if (threadIdx.x == 25) wl[25] = bias[c];
    __syncthreads();

    const float* ip = in  + (long)bc * HW_;
    float*       op = out + (long)bc * HW_;
    const int x = threadIdx.x, y = chunk;
    if (y + 3 < 256) __builtin_prefetch(ip + (y + 3) * 256 + x, 0, 1);

    float s = wl[25];
#pragma unroll
    for (int dy = 0; dy < 5; ++dy) {
        int yy = y + dy - 2;
        if (yy < 0 || yy > 255) continue;
#pragma unroll
        for (int dx = 0; dx < 5; ++dx) {
            int xx = x + dx - 2;
            if (xx < 0 || xx > 255) continue;
            s += wl[dy * 5 + dx] * ip[yy * 256 + xx];
        }
    }
    op[y * 256 + x] = s;
}

// ---------------------------------------------------------------------------
// Depthwise 3x3, dilation 3, pad 3.
// ---------------------------------------------------------------------------
__global__ __launch_bounds__(256)
void dw_conv3d3(const float* __restrict__ in, const float* __restrict__ w,
                const float* __restrict__ bias, float* __restrict__ out, int Cn)
{
    const int chunk = blockIdx.x & 255;
    const int bc    = blockIdx.x >> 8;
    const int c     = bc % Cn;

    __shared__ float wl[10];
    if (threadIdx.x < 9) wl[threadIdx.x] = w[c * 9 + threadIdx.x];
    if (threadIdx.x == 9) wl[9] = bias[c];
    __syncthreads();

    const float* ip = in  + (long)bc * HW_;
    float*       op = out + (long)bc * HW_;
    const int x = threadIdx.x, y = chunk;
    if (y + 4 < 256) __builtin_prefetch(ip + (y + 4) * 256 + x, 0, 1);

    float s = wl[9];
#pragma unroll
    for (int dy = 0; dy < 3; ++dy) {
        int yy = y + (dy - 1) * 3;
        if (yy < 0 || yy > 255) continue;
#pragma unroll
        for (int dx = 0; dx < 3; ++dx) {
            int xx = x + (dx - 1) * 3;
            if (xx < 0 || xx > 255) continue;
            s += wl[dy * 3 + dx] * ip[yy * 256 + xx];
        }
    }
    op[y * 256 + x] = s;
}

// ---------------------------------------------------------------------------
// GroupNorm(2 groups) statistics: per (b, g) sum and sumsq via atomics.
// ---------------------------------------------------------------------------
__global__ __launch_bounds__(256)
void gn_stats(const float* __restrict__ x, float* __restrict__ stats, int Cn)
{
    long idx = (long)blockIdx.x * 256 + threadIdx.x;
    int bc = (int)(idx >> 16);
    int c = bc % Cn, b = bc / Cn;
    int g = (c >= (Cn >> 1)) ? 1 : 0;
    float v = x[idx];
    __shared__ float s1[256], s2[256];
    s1[threadIdx.x] = v; s2[threadIdx.x] = v * v;
    __syncthreads();
    for (int off = 128; off > 0; off >>= 1) {
        if (threadIdx.x < off) {
            s1[threadIdx.x] += s1[threadIdx.x + off];
            s2[threadIdx.x] += s2[threadIdx.x + off];
        }
        __syncthreads();
    }
    if (threadIdx.x == 0) {
        atomicAdd(&stats[(b * 2 + g) * 2 + 0], s1[0]);
        atomicAdd(&stats[(b * 2 + g) * 2 + 1], s2[0]);
    }
}

// ---------------------------------------------------------------------------
// GroupNorm apply + leaky ReLU(0.2), in place.
// ---------------------------------------------------------------------------
__global__ __launch_bounds__(256)
void gn_apply(float* __restrict__ x, const float* __restrict__ stats,
              const float* __restrict__ gamma, const float* __restrict__ beta, int Cn)
{
    long idx = (long)blockIdx.x * 256 + threadIdx.x;
    int bc = (int)(idx >> 16);
    int c = bc % Cn, b = bc / Cn;
    int g = (c >= (Cn >> 1)) ? 1 : 0;
    float N  = (float)((Cn >> 1) * (long)HW_);
    float su = stats[(b * 2 + g) * 2 + 0];
    float ss = stats[(b * 2 + g) * 2 + 1];
    float mu  = su / N;
    float var = ss / N - mu * mu;
    float inv = rsqrtf(var + 1e-5f);
    float v = (x[idx] - mu) * inv * gamma[c] + beta[c];
    x[idx] = v > 0.f ? v : 0.2f * v;
}

// ---------------------------------------------------------------------------
// Row L2 norms over HW: nrm[b*64+c] = sqrt(sum_p src[b,c,p]^2)
// ---------------------------------------------------------------------------
__global__ __launch_bounds__(256)
void rownorm(const float* __restrict__ src, float* __restrict__ nrm, int batchCh)
{
    int row = blockIdx.x;
    int b = row >> 6, c = row & 63;
    const float* p = src + (long)b * batchCh * HW_ + (long)c * HW_;
    float s = 0.f;
    for (int i = threadIdx.x; i < HW_; i += 256) { float v = p[i]; s += v * v; }
    __shared__ float red[256];
    red[threadIdx.x] = s; __syncthreads();
    for (int off = 128; off > 0; off >>= 1) {
        if (threadIdx.x < off) red[threadIdx.x] += red[threadIdx.x + off];
        __syncthreads();
    }
    if (threadIdx.x == 0) nrm[row] = sqrtf(red[0]);
}

// ---------------------------------------------------------------------------
// Channel-attention gram matrices via f32 WMMA 16x16x4 over K = HW.
// Two interleaved accumulators break the WMMA->WMMA dependency chain.
// ---------------------------------------------------------------------------
__global__ __launch_bounds__(256)
void attn_dot_wmma(const float* __restrict__ q, const float* __restrict__ k,
                   float* __restrict__ attn_raw)
{
    const int WGS = 32;                        // workgroups per (b,h)
    int bh = blockIdx.x / WGS;
    int wg = blockIdx.x % WGS;
    int b = bh >> 2, h = bh & 3;

    const int tid = threadIdx.x, lane = tid & 31, ln = lane & 15, hi = lane >> 4;
    const float* qrow = q + (long)b * 64  * HW_ + (long)(h * 16 + ln) * HW_;
    const float* krow = k + (long)b * 128 * HW_ + (long)(h * 16 + ln) * HW_;
    const int koff = hi ? 2 : 0;

    const int stride = WGS * 8;                // 256 waves per (b,h)
    int chunk0 = wg * 8 + (tid >> 5);          // in [0, 256)

    // 16384 chunks total, 64 per wave -> exactly 32 guard-free pair iterations
    v8f c0 = {}, c1 = {};
    for (int ch = chunk0; ch < (HW_ >> 2); ch += 2 * stride) {
        int p0 = (ch << 2) + koff;
        v2f a0  = *(const v2f*)(qrow + p0);
        v2f bv0 = *(const v2f*)(krow + p0);
        c0 = __builtin_amdgcn_wmma_f32_16x16x4_f32(false, a0, false, bv0,
                                                   (short)0, c0, false, false);
        int p1 = ((ch + stride) << 2) + koff;
        v2f a1  = *(const v2f*)(qrow + p1);
        v2f bv1 = *(const v2f*)(krow + p1);
        c1 = __builtin_amdgcn_wmma_f32_16x16x4_f32(false, a1, false, bv1,
                                                   (short)0, c1, false, false);
    }
#pragma unroll
    for (int r = 0; r < 8; ++r) {
        int m = r + (hi ? 8 : 0);
        atomicAdd(&attn_raw[(long)bh * 256 + m * 16 + ln], c0[r] + c1[r]);
    }
}

// ---------------------------------------------------------------------------
// Fold L2 norms + temperature, softmax over last dim (16).
// ---------------------------------------------------------------------------
__global__ __launch_bounds__(256)
void softmax_attn(const float* __restrict__ raw, const float* __restrict__ nq,
                  const float* __restrict__ nk, const float* __restrict__ temp,
                  float* __restrict__ outp)
{
    int bh = blockIdx.x; int b = bh >> 2, h = bh & 3;
    int tid = threadIdx.x; int i = tid >> 4, j = tid & 15;
    float qn = fmaxf(nq[b * 64 + h * 16 + i], 1e-12f);
    float kn = fmaxf(nk[b * 64 + h * 16 + j], 1e-12f);
    float l = raw[bh * 256 + tid] / (qn * kn) * temp[h];
    __shared__ float s[256];
    s[tid] = l; __syncthreads();
    float mx = -1e30f;
    for (int jj = 0; jj < 16; ++jj) mx = fmaxf(mx, s[i * 16 + jj]);
    float e = __expf(l - mx);
    __syncthreads(); s[tid] = e; __syncthreads();
    float sum = 0.f;
    for (int jj = 0; jj < 16; ++jj) sum += s[i * 16 + jj];
    outp[bh * 256 + tid] = e / sum;
}

// ---------------------------------------------------------------------------
// M[b] = proj_w (64x64) @ blockdiag(attn[b])  -> fused (attn@v) + proj matrix
// ---------------------------------------------------------------------------
__global__ __launch_bounds__(256)
void combine_proj(const float* __restrict__ attn, const float* __restrict__ P,
                  float* __restrict__ M)
{
    int b = blockIdx.x;
    __shared__ float a[1024];
    for (int i = threadIdx.x; i < 1024; i += 256) a[i] = attn[b * 1024 + i];
    __syncthreads();
    for (int idx = threadIdx.x; idx < 4096; idx += 256) {
        int co = idx >> 6, dg = idx & 63;
        int h = dg >> 4, dj = dg & 15;
        float s = 0.f;
#pragma unroll
        for (int cp = 0; cp < 16; ++cp)
            s += P[co * 64 + h * 16 + cp] * a[h * 256 + cp * 16 + dj];
        M[b * 4096 + idx] = s;
    }
}

__global__ __launch_bounds__(256)
void zero_ws(float* p, int n)
{
    int i = blockIdx.x * 256 + threadIdx.x;
    if (i < n) p[i] = 0.f;
}

// ---------------------------------------------------------------------------
extern "C" void kernel_launch(void* const* d_in, const int* in_sizes, int n_in,
                              void* d_out, int out_size, void* d_ws, size_t ws_size,
                              hipStream_t stream)
{
    const float* x       = (const float*)d_in[0];
    const float* y       = (const float*)d_in[1];
    const float* kv_w    = (const float*)d_in[2];
    const float* q_w     = (const float*)d_in[3];
    const float* proj_w  = (const float*)d_in[4];
    const float* kv_c0_w = (const float*)d_in[5];
    const float* kv_c0_b = (const float*)d_in[6];
    const float* kv_cs_w = (const float*)d_in[7];
    const float* kv_cs_b = (const float*)d_in[8];
    const float* kv_c1_w = (const float*)d_in[9];
    const float* kv_c1_b = (const float*)d_in[10];
    const float* kv_gn_g = (const float*)d_in[11];
    const float* kv_gn_b = (const float*)d_in[12];
    const float* q_c0_w  = (const float*)d_in[13];
    const float* q_c0_b  = (const float*)d_in[14];
    const float* q_cs_w  = (const float*)d_in[15];
    const float* q_cs_b  = (const float*)d_in[16];
    const float* q_c1_w  = (const float*)d_in[17];
    const float* q_c1_b  = (const float*)d_in[18];
    const float* q_gn_g  = (const float*)d_in[19];
    const float* q_gn_b  = (const float*)d_in[20];
    const float* temp    = (const float*)d_in[21];
    float* out = (float*)d_out;

    const long HW = HW_;
    float* ws   = (float*)d_ws;
    float* bufA = ws;                      // 4*128*HW floats (128 MB)
    float* bufB = ws + 33554432L;          // 4*128*HW floats (128 MB)
    float* bufC = bufA;                    // q path reuses bufA region (64 MB)
    float* bufD = bufA + 16777216L;        // (64 MB)
    float* smallr  = ws + 67108864L;
    float* kvstats = smallr;               // 16
    float* qstats  = smallr + 16;          // 16
    float* nq      = smallr + 32;          // 256
    float* nk      = smallr + 288;         // 256
    float* rawA    = smallr + 544;         // 4096
    float* softA   = smallr + 4640;        // 4096
    float* Mmat    = smallr + 8736;        // 16384

    // zero stats + attn accumulators
    zero_ws<<<(4640 + 255) / 256, 256, 0, stream>>>(smallr, 4640);

    // ---- kv branch ----
    pw_gemm_wmma<<<2048, 256, 0, stream>>>(y, kv_w, nullptr, bufA, 64, 128,
                                           64L * HW, 128L * HW, 0L);
    dw_conv5 <<<4 * 128 * 256, 256, 0, stream>>>(bufA, kv_c0_w, kv_c0_b, bufB, 128);
    dw_conv3d3<<<4 * 128 * 256, 256, 0, stream>>>(bufB, kv_cs_w, kv_cs_b, bufA, 128);
    pw_gemm_wmma<<<2048, 256, 0, stream>>>(bufA, kv_c1_w, kv_c1_b, bufB, 128, 128,
                                           128L * HW, 128L * HW, 0L);
    gn_stats<<<4 * 128 * 256, 256, 0, stream>>>(bufB, kvstats, 128);
    gn_apply<<<4 * 128 * 256, 256, 0, stream>>>(bufB, kvstats, kv_gn_g, kv_gn_b, 128);

    // ---- q branch ----
    pw_gemm_wmma<<<2048, 256, 0, stream>>>(x, q_w, nullptr, bufC, 64, 64,
                                           64L * HW, 64L * HW, 0L);
    dw_conv5 <<<4 * 64 * 256, 256, 0, stream>>>(bufC, q_c0_w, q_c0_b, bufD, 64);
    dw_conv3d3<<<4 * 64 * 256, 256, 0, stream>>>(bufD, q_cs_w, q_cs_b, bufC, 64);
    pw_gemm_wmma<<<2048, 256, 0, stream>>>(bufC, q_c1_w, q_c1_b, bufD, 64, 64,
                                           64L * HW, 64L * HW, 0L);
    gn_stats<<<4 * 64 * 256, 256, 0, stream>>>(bufD, qstats, 64);
    gn_apply<<<4 * 64 * 256, 256, 0, stream>>>(bufD, qstats, q_gn_g, q_gn_b, 64);

    // ---- attention ----
    rownorm<<<256, 256, 0, stream>>>(bufD, nq, 64);   // q norms
    rownorm<<<256, 256, 0, stream>>>(bufB, nk, 128);  // k norms (k = kv[:, :64])
    attn_dot_wmma<<<16 * 32, 256, 0, stream>>>(bufD, bufB, rawA);
    softmax_attn<<<16, 256, 0, stream>>>(rawA, nq, nk, temp, softA);
    combine_proj<<<4, 256, 0, stream>>>(softA, proj_w, Mmat);

    // ---- fused (attn @ v) + proj as one 64x64 GEMM per batch ----
    pw_gemm_wmma<<<2048, 256, 0, stream>>>(bufB + 64L * HW, Mmat, nullptr, out,
                                           64, 64, 128L * HW, 64L * HW, 4096L);
}
